// PCT_patch_semseg_25366076850321
// MI455X (gfx1250) — compile-verified, matrix operand order checked by reading, not run on hardware
//
#include <hip/hip_runtime.h>
#include <math.h>

// ---------------------------------------------------------------------------
// PCT semantic segmentation pipeline for MI455X (gfx1250, wave32, WMMA).
// Heavy GEMMs -> v_wmma_f32_16x16x32_f16 (f16 in, f32 accum).
// A-tile staging -> Tensor Data Mover (tensor_load_to_lds + s_wait_tensorcnt)
// with manual-staging fallback if the builtin is unavailable.
// ---------------------------------------------------------------------------

#define BB   8
#define NN   2048
#define KNN  32
#define EPSF 1e-5f

typedef _Float16 half8 __attribute__((ext_vector_type(8)));
typedef _Float16 v16h  __attribute__((ext_vector_type(16)));
typedef float    v8f   __attribute__((ext_vector_type(8)));

typedef unsigned int v4u __attribute__((ext_vector_type(4)));
typedef int          v8i __attribute__((ext_vector_type(8)));
typedef int          v4i __attribute__((ext_vector_type(4)));

#if __has_builtin(__builtin_amdgcn_tensor_load_to_lds)
#define PCT_USE_TDM 1
#else
#define PCT_USE_TDM 0
#endif

// ---------------------------------------------------------------------------
// Generic WMMA GEMM: C[b] = A[b] (MxK, row-major, ld=lda) * B[b] (KxN, ld=ldb)
//                    (+ per-row bias).  M%128==0, N%128==0, K%32==0.
// Block = 256 threads = 8 waves; tile = 128x128; wave w owns rows 16w..16w+15.
// A tile (128x32 f32, stride lda) is DMA'd to LDS by the TDM (wave 0 issues
// the descriptor, waits TENSORcnt, workgroup barrier publishes it); B tile is
// staged transposed + f16-converted by all waves.
// ---------------------------------------------------------------------------
__launch_bounds__(256)
__global__ void gemm_wmma_f16(const float* __restrict__ A, long strideA, int lda,
                              const float* __restrict__ Bm, long strideB, int ldb,
                              float* __restrict__ C, long strideC, int ldc,
                              const float* __restrict__ bias, long strideBias,
                              int M, int Nc, int Kc)
{
#if PCT_USE_TDM
    __shared__ __attribute__((aligned(16))) float    As32[128 * 32]; // [row][k] f32 (TDM dest)
#else
    __shared__ __attribute__((aligned(16))) _Float16 As[128 * 32];   // [row][k] f16
#endif
    __shared__ __attribute__((aligned(16))) _Float16 Bs[128 * 32];   // [col][k] (transposed)

    const int b = blockIdx.z;
    A  += (long)b * strideA;
    Bm += (long)b * strideB;
    C  += (long)b * strideC;

    const int tid  = threadIdx.x;
    const int wave = tid >> 5;
    const int lane = tid & 31;
    const int mL   = lane & 15;
    const int g    = lane >> 4;
    const int row0 = blockIdx.y * 128;
    const int col0 = blockIdx.x * 128;

    v8f acc[8];
    #pragma unroll
    for (int i = 0; i < 8; i++)
        #pragma unroll
        for (int j = 0; j < 8; j++) acc[i][j] = 0.0f;

    const int ar  = tid >> 1;           // A: row, 2 threads per row (fallback path)
    const int ac0 = (tid & 1) * 16;     // A: 16 consecutive k per thread
    const int bk  = tid >> 3;           // B: k row
    const int bn0 = (tid & 7) * 16;     // B: 16 consecutive cols per thread

    for (int kt = 0; kt < Kc; kt += 32) {
#if PCT_USE_TDM
        // ---- A tile via Tensor Data Mover: 2D tile 32(x) x 128(y) f32 ----
        if (tid < 32) {
            const float* gA = A + (long)row0 * lda + kt;
            unsigned long long ga = (unsigned long long)(size_t)gA;
            unsigned ldsa = (unsigned)(size_t)(void*)As32;
            // D# group0: count=1 | lds_addr | global_addr[56:0] | type=2
            v4u g0 = { 1u, ldsa,
                       (unsigned)(ga & 0xffffffffull),
                       (unsigned)((ga >> 32) & 0x1ffffffull) | 0x80000000u };
            // D# group1: data_size=4B; tensor_dim0=32, tensor_dim1=128;
            //            tile_dim0=32, tile_dim1=128; dim0 stride = lda
            v8i g1 = { (int)(2u << 16),
                       (int)(32u << 16),
                       (int)(128u << 16),
                       (int)(32u << 16),
                       (int)128,
                       (int)(unsigned)lda,
                       0, 0 };
            v4i gz = { 0, 0, 0, 0 };
#if defined(__clang_major__) && (__clang_major__ >= 23)
            v8i gz8 = { 0, 0, 0, 0, 0, 0, 0, 0 };
            __builtin_amdgcn_tensor_load_to_lds(g0, g1, gz, gz, gz8, 0);
#else
            __builtin_amdgcn_tensor_load_to_lds(g0, g1, gz, gz, 0);
#endif
        }
#else
        // ---- A tile manual staging (f32 -> f16) ----
        {
            const float* src = A + (long)(row0 + ar) * lda + kt + ac0;
            #pragma unroll
            for (int i = 0; i < 16; i++) As[ar * 32 + ac0 + i] = (_Float16)src[i];
        }
#endif
        // ---- B tile (32x128) staged transposed into LDS as f16: Bs[col*32+k] ----
        {
            const float* src = Bm + (long)(kt + bk) * ldb + col0 + bn0;
            #pragma unroll
            for (int i = 0; i < 16; i++) Bs[(bn0 + i) * 32 + bk] = (_Float16)src[i];
        }
        // Prefetch next B K-tile while we compute this one
        if (kt + 32 < Kc) {
            __builtin_prefetch(Bm + (long)(kt + 32 + bk) * ldb + col0 + bn0, 0, 1);
#if !PCT_USE_TDM
            __builtin_prefetch(A + (long)(row0 + ar) * lda + kt + 32 + ac0, 0, 1);
#endif
        }
#if PCT_USE_TDM
        if (tid < 32) __builtin_amdgcn_s_wait_tensorcnt(0);
#endif
        __syncthreads();

        // A fragment: lane L holds row (L&15); halves 0..7 -> k=8g+0..7,
        // halves 8..15 -> k=16+8g+0..7  (CDNA5 16-bit A 16x32 layout)
        v16h afrag;
        {
#if PCT_USE_TDM
            const float* ap = &As32[(wave * 16 + mL) * 32 + 8 * g];
            #pragma unroll
            for (int i = 0; i < 8; i++) {
                afrag[i]     = (_Float16)ap[i];
                afrag[i + 8] = (_Float16)ap[i + 16];
            }
#else
            const _Float16* ap = &As[(wave * 16 + mL) * 32 + 8 * g];
            half8 lo = *(const half8*)ap;
            half8 hi = *(const half8*)(ap + 16);
            #pragma unroll
            for (int i = 0; i < 8; i++) { afrag[i] = lo[i]; afrag[i + 8] = hi[i]; }
#endif
        }
        #pragma unroll
        for (int ct = 0; ct < 8; ct++) {
            v16h bfrag;
            const _Float16* bp = &Bs[(ct * 16 + mL) * 32 + 8 * g];
            half8 lo = *(const half8*)bp;
            half8 hi = *(const half8*)(bp + 16);
            #pragma unroll
            for (int i = 0; i < 8; i++) { bfrag[i] = lo[i]; bfrag[i + 8] = hi[i]; }
            acc[ct] = __builtin_amdgcn_wmma_f32_16x16x32_f16(
                false, afrag, false, bfrag, (short)0, acc[ct], false, false);
        }
        __syncthreads();
    }

    // C/D layout: VGPR r, lane L -> row r + 8*(L>>4), col L&15
    const float* bptr = bias ? (bias + (long)b * strideBias) : nullptr;
    #pragma unroll
    for (int ct = 0; ct < 8; ct++) {
        #pragma unroll
        for (int r = 0; r < 8; r++) {
            int grow = row0 + wave * 16 + r + 8 * g;
            int gcol = col0 + ct * 16 + mL;
            float v = acc[ct][r];
            if (bptr) v += bptr[grow];
            C[(long)grow * ldc + gcol] = v;
        }
    }
}

// ---------------------------------------------------------------------------
// kNN: block per (b, n); distances to all N points in LDS, 32x iterative
// argmin (ties -> smaller index, matching top_k of negative distance).
// ---------------------------------------------------------------------------
__launch_bounds__(256)
__global__ void knn_kernel(const float* __restrict__ x, int C, int* __restrict__ idx)
{
    __shared__ float center[64];
    __shared__ float d[NN];
    __shared__ float rv[256];
    __shared__ int   ri[256];
    const int b = blockIdx.x / NN;
    const int n = blockIdx.x % NN;
    const int tid = threadIdx.x;
    const float* xb = x + (long)b * C * NN;
    if (tid < C) center[tid] = xb[tid * NN + n];
    __syncthreads();
    for (int m = tid; m < NN; m += 256) {
        float s = 0.f;
        for (int c = 0; c < C; c++) {
            float df = xb[c * NN + m] - center[c];
            s += df * df;
        }
        d[m] = s;
    }
    __syncthreads();
    int* out = idx + ((long)b * NN + n) * KNN;
    for (int it = 0; it < KNN; it++) {
        float best = 3.4e38f; int bi = NN;
        for (int m = tid; m < NN; m += 256) {
            float v = d[m];
            if (v < best || (v == best && m < bi)) { best = v; bi = m; }
        }
        rv[tid] = best; ri[tid] = bi;
        __syncthreads();
        for (int s2 = 128; s2 > 0; s2 >>= 1) {
            if (tid < s2) {
                float ov = rv[tid + s2]; int oi = ri[tid + s2];
                if (ov < rv[tid] || (ov == rv[tid] && oi < ri[tid])) { rv[tid] = ov; ri[tid] = oi; }
            }
            __syncthreads();
        }
        if (tid == 0) { out[it] = ri[0]; d[ri[0]] = 3.4e38f; }
        __syncthreads();
    }
}

// ---------------------------------------------------------------------------
// EdgeConv1: 6 -> 64; per (b,n): e[o,k]; keep max over k + per-(b,n) channel
// partial sum/sumsq (deterministic two-stage stats, no float atomics).
// ---------------------------------------------------------------------------
__launch_bounds__(256)
__global__ void edge1_kernel(const float* __restrict__ x, const int* __restrict__ idx,
                             const float* __restrict__ W, const float* __restrict__ bias,
                             float* __restrict__ emax,
                             float* __restrict__ epsum, float* __restrict__ epsq)
{
    __shared__ float Ws[64 * 6];
    __shared__ float bs[64];
    __shared__ float feat[KNN * 6];
    __shared__ float rmax[256], rsum[256], rsq[256];
    const int b = blockIdx.x / NN, n = blockIdx.x % NN;
    const int tid = threadIdx.x;
    for (int i = tid; i < 64 * 6; i += 256) Ws[i] = W[i];
    if (tid < 64) bs[tid] = bias[tid];
    const int* ids = idx + ((long)b * NN + n) * KNN;
    const float* xb = x + (long)b * 3 * NN;
    if (tid < KNN) {
        int nb = ids[tid];
        for (int c = 0; c < 3; c++) {
            float cv = xb[c * NN + n];
            float nv = xb[c * NN + nb];
            feat[tid * 6 + c] = nv - cv;
            feat[tid * 6 + 3 + c] = cv;
        }
    }
    __syncthreads();
    const int o = tid & 63, grp = tid >> 6;
    float lmax = -3.4e38f, ls = 0.f, lsq = 0.f;
    for (int j = 0; j < 8; j++) {
        int k = grp * 8 + j;
        float e = bs[o];
        for (int c = 0; c < 6; c++) e += Ws[o * 6 + c] * feat[k * 6 + c];
        lmax = fmaxf(lmax, e); ls += e; lsq += e * e;
    }
    rmax[tid] = lmax; rsum[tid] = ls; rsq[tid] = lsq;
    __syncthreads();
    if (tid < 64) {
        for (int g2 = 1; g2 < 4; g2++) {
            lmax = fmaxf(lmax, rmax[tid + 64 * g2]);
            ls  += rsum[tid + 64 * g2];
            lsq += rsq[tid + 64 * g2];
        }
        emax[((long)b * 64 + o) * NN + n] = lmax;
        epsum[((long)b * NN + n) * 64 + o] = ls;
        epsq [((long)b * NN + n) * 64 + o] = lsq;
    }
}

// EdgeConv2: 128 -> 64
__launch_bounds__(256)
__global__ void edge2_kernel(const float* __restrict__ f1, const int* __restrict__ idx,
                             const float* __restrict__ W, const float* __restrict__ bias,
                             float* __restrict__ emax,
                             float* __restrict__ epsum, float* __restrict__ epsq)
{
    __shared__ float Ws[64 * 128];
    __shared__ float bs[64];
    __shared__ float feat[KNN * 128];
    __shared__ float rmax[256], rsum[256], rsq[256];
    const int b = blockIdx.x / NN, n = blockIdx.x % NN;
    const int tid = threadIdx.x;
    for (int i = tid; i < 64 * 128; i += 256) Ws[i] = W[i];
    if (tid < 64) bs[tid] = bias[tid];
    const int* ids = idx + ((long)b * NN + n) * KNN;
    const float* fb = f1 + (long)b * 64 * NN;
    for (int q = tid; q < KNN * 128; q += 256) {
        int k = q >> 7, c = q & 127;
        int nb = ids[k];
        float v;
        if (c < 64) v = fb[c * NN + nb] - fb[c * NN + n];
        else        v = fb[(c - 64) * NN + n];
        feat[q] = v;
    }
    __syncthreads();
    const int o = tid & 63, grp = tid >> 6;
    float lmax = -3.4e38f, ls = 0.f, lsq = 0.f;
    for (int j = 0; j < 8; j++) {
        int k = grp * 8 + j;
        float e = bs[o];
        const float* wo = &Ws[o * 128];
        const float* fk = &feat[k * 128];
        for (int c = 0; c < 128; c++) e += wo[c] * fk[c];
        lmax = fmaxf(lmax, e); ls += e; lsq += e * e;
    }
    rmax[tid] = lmax; rsum[tid] = ls; rsq[tid] = lsq;
    __syncthreads();
    if (tid < 64) {
        for (int g2 = 1; g2 < 4; g2++) {
            lmax = fmaxf(lmax, rmax[tid + 64 * g2]);
            ls  += rsum[tid + 64 * g2];
            lsq += rsq[tid + 64 * g2];
        }
        emax[((long)b * 64 + o) * NN + n] = lmax;
        epsum[((long)b * NN + n) * 64 + o] = ls;
        epsq [((long)b * NN + n) * 64 + o] = lsq;
    }
}

// Reduce edge per-(b,n) channel partials -> channel sum/sumsq (64 channels)
__launch_bounds__(256)
__global__ void epart_reduce_kernel(const float* __restrict__ ep, const float* __restrict__ eq,
                                    float* __restrict__ ssum, float* __restrict__ ssq)
{
    __shared__ float rs[256], rq[256];
    const int o = blockIdx.x;
    const int tid = threadIdx.x;
    float s = 0.f, q = 0.f;
    for (long r = tid; r < (long)BB * NN; r += 256) { s += ep[r * 64 + o]; q += eq[r * 64 + o]; }
    rs[tid] = s; rq[tid] = q;
    __syncthreads();
    for (int s2 = 128; s2 > 0; s2 >>= 1) {
        if (tid < s2) { rs[tid] += rs[tid + s2]; rq[tid] += rq[tid + s2]; }
        __syncthreads();
    }
    if (tid == 0) { ssum[o] = rs[0]; ssq[o] = rq[0]; }
}

// In-place BatchNorm (train stats) + ReLU over [B, C, N] layout
__global__ void bn_relu_kernel(float* __restrict__ buf, const float* __restrict__ ssum,
                               const float* __restrict__ ssq, int C, float invCount, long total)
{
    long i = (long)blockIdx.x * blockDim.x + threadIdx.x;
    if (i >= total) return;
    int c = (int)((i / NN) % C);
    float mean = ssum[c] * invCount;
    float var  = ssq[c] * invCount - mean * mean;
    float v = (buf[i] - mean) * rsqrtf(var + EPSF);
    buf[i] = v > 0.f ? v : 0.f;
}

// Channel sum/sumsq over [B, C, N], one block per channel (deterministic)
__launch_bounds__(256)
__global__ void chan_stats_kernel(const float* __restrict__ buf, int C,
                                  float* __restrict__ ssum, float* __restrict__ ssq)
{
    __shared__ float rs[256], rq[256];
    const int c = blockIdx.x;
    const int tid = threadIdx.x;
    float s = 0.f, q = 0.f;
    for (int b = 0; b < BB; b++) {
        const float* p = buf + ((long)b * C + c) * NN;
        for (int n = tid; n < NN; n += 256) { float v = p[n]; s += v; q += v * v; }
    }
    rs[tid] = s; rq[tid] = q;
    __syncthreads();
    for (int s2 = 128; s2 > 0; s2 >>= 1) {
        if (tid < s2) { rs[tid] += rs[tid + s2]; rq[tid] += rq[tid + s2]; }
        __syncthreads();
    }
    if (tid == 0) { ssum[c] = rs[0]; ssq[c] = rq[0]; }
}

// feat = concat(f1, f2) along channel -> [B,128,N]
__global__ void concat_kernel(const float* __restrict__ f1, const float* __restrict__ f2,
                              float* __restrict__ feat, long total)
{
    long i = (long)blockIdx.x * blockDim.x + threadIdx.x;
    if (i >= total) return;
    int n = (int)(i % NN); long t = i / NN; int c = (int)(t % 128); int b = (int)(t / 128);
    feat[i] = (c < 64) ? f1[((long)b * 64 + c) * NN + n]
                       : f2[((long)b * 64 + (c - 64)) * NN + n];
}

// xq[B,N,32] and xk[B,32,N] from xs [B(stride),128,N]
__launch_bounds__(256)
__global__ void qk_kernel(const float* __restrict__ xs, long xsStride,
                          const float* __restrict__ Wq, const float* __restrict__ Wk,
                          float* __restrict__ xq, float* __restrict__ xk)
{
    __shared__ float Wqs[32 * 128], Wks[32 * 128];
    __shared__ float xt[8 * 128];
    const int b  = blockIdx.x / (NN / 8);
    const int n0 = (blockIdx.x % (NN / 8)) * 8;
    const int tid = threadIdx.x;
    for (int i = tid; i < 4096; i += 256) { Wqs[i] = Wq[i]; Wks[i] = Wk[i]; }
    const float* xb = xs + (long)b * xsStride;
    for (int i = tid; i < 8 * 128; i += 256) {
        int nl = i >> 7, c = i & 127;
        xt[nl * 128 + c] = xb[(long)c * NN + n0 + nl];
    }
    __syncthreads();
    const int q = tid & 31, nl = tid >> 5;
    float aq = 0.f, ak = 0.f;
    const float* wq = &Wqs[q * 128];
    const float* wk = &Wks[q * 128];
    const float* xv = &xt[nl * 128];
    for (int c = 0; c < 128; c++) { float v = xv[c]; aq += wq[c] * v; ak += wk[c] * v; }
    const int n = n0 + nl;
    xq[((long)b * NN + n) * 32 + q] = aq;
    xk[(long)b * 32 * NN + (long)q * NN + n] = ak;
}

// Row softmax over attn[b, n, :]
__launch_bounds__(256)
__global__ void softmax_kernel(float* __restrict__ attn)
{
    __shared__ float red[256];
    float* p = attn + (long)blockIdx.x * NN;
    const int tid = threadIdx.x;
    float m = -3.4e38f;
    for (int i = tid; i < NN; i += 256) m = fmaxf(m, p[i]);
    red[tid] = m; __syncthreads();
    for (int s = 128; s > 0; s >>= 1) { if (tid < s) red[tid] = fmaxf(red[tid], red[tid + s]); __syncthreads(); }
    m = red[0]; __syncthreads();
    float s = 0.f;
    for (int i = tid; i < NN; i += 256) { float e = __expf(p[i] - m); p[i] = e; s += e; }
    red[tid] = s; __syncthreads();
    for (int s2 = 128; s2 > 0; s2 >>= 1) { if (tid < s2) red[tid] += red[tid + s2]; __syncthreads(); }
    float inv = 1.f / red[0];
    for (int i = tid; i < NN; i += 256) p[i] *= inv;
}

// Column sums over n, then scale: attn[:,m] /= (1e-6 + colsum[m])
__launch_bounds__(256)
__global__ void colscale_kernel(float* __restrict__ attn)
{
    const int b = blockIdx.x >> 3;
    const int m = (blockIdx.x & 7) * 256 + threadIdx.x;
    float* p = attn + (long)b * NN * NN + m;
    float s = 0.f;
    for (int n = 0; n < NN; n++) s += p[(long)n * NN];
    const float inv = 1.f / (1e-6f + s);
    for (int n = 0; n < NN; n++) p[(long)n * NN] *= inv;
}

// u = x - xr (in place into xr)
__global__ void sub_kernel(const float* __restrict__ xs, long xsStride,
                           float* __restrict__ xr, long total)
{
    long i = (long)blockIdx.x * blockDim.x + threadIdx.x;
    if (i >= total) return;
    int n = (int)(i % NN); long t = i / NN; int c = (int)(t % 128); int b = (int)(t / 128);
    xr[i] = xs[(long)b * xsStride + (long)c * NN + n] - xr[i];
}

// xs_out (cat slice) = xs + relu(BN(t))
__global__ void sa_out_kernel(const float* __restrict__ xs, long xsStride,
                              const float* __restrict__ t,
                              const float* __restrict__ ssum, const float* __restrict__ ssq,
                              float invCount, float* __restrict__ catOut, long total)
{
    long i = (long)blockIdx.x * blockDim.x + threadIdx.x;
    if (i >= total) return;
    int n = (int)(i % NN); long tt = i / NN; int c = (int)(tt % 128); int b = (int)(tt / 128);
    float mean = ssum[c] * invCount;
    float var  = ssq[c] * invCount - mean * mean;
    float v = (t[i] - mean) * rsqrtf(var + EPSF);
    v = v > 0.f ? v : 0.f;
    catOut[(long)b * 512 * NN + (long)c * NN + n] =
        xs[(long)b * xsStride + (long)c * NN + n] + v;
}

// Fuse stats: per (b,c) sum/sumsq/max over n of y[B,1024,N]
__launch_bounds__(256)
__global__ void ystats_kernel(const float* __restrict__ y,
                              float* __restrict__ psum, float* __restrict__ psq,
                              float* __restrict__ pmax)
{
    __shared__ float rs[256], rq[256], rm[256];
    const int bc = blockIdx.x;
    const float* p = y + (long)bc * NN;
    const int tid = threadIdx.x;
    float s = 0.f, q = 0.f, m = -3.4e38f;
    for (int n = tid; n < NN; n += 256) { float v = p[n]; s += v; q += v * v; m = fmaxf(m, v); }
    rs[tid] = s; rq[tid] = q; rm[tid] = m;
    __syncthreads();
    for (int s2 = 128; s2 > 0; s2 >>= 1) {
        if (tid < s2) {
            rs[tid] += rs[tid + s2]; rq[tid] += rq[tid + s2];
            rm[tid] = fmaxf(rm[tid], rm[tid + s2]);
        }
        __syncthreads();
    }
    if (tid == 0) { psum[bc] = rs[0]; psq[bc] = rq[0]; pmax[bc] = rm[0]; }
}

// g[b,c] = relu(BN(max_n y))  (BN monotone -> max commutes)
__global__ void g_kernel(const float* __restrict__ psum, const float* __restrict__ psq,
                         const float* __restrict__ pmax, float* __restrict__ g)
{
    int c = blockIdx.x * blockDim.x + threadIdx.x;
    if (c >= 1024) return;
    float s = 0.f, q = 0.f;
    for (int b = 0; b < BB; b++) { s += psum[b * 1024 + c]; q += psq[b * 1024 + c]; }
    const float invc = 1.f / ((float)BB * (float)NN);
    float mean = s * invc, var = q * invc - mean * mean;
    float rsd = rsqrtf(var + EPSF);
    for (int b = 0; b < BB; b++) {
        float v = (pmax[b * 1024 + c] - mean) * rsd;
        g[b * 1024 + c] = v > 0.f ? v : 0.f;
    }
}

// ghb[b,o] = bc1[o] + Wc1[o, :1024] . g[b]   (global-feature half of Wc1@h)
__launch_bounds__(256)
__global__ void gh_kernel(const float* __restrict__ Wc1, const float* __restrict__ bc1,
                          const float* __restrict__ g, float* __restrict__ ghb)
{
    __shared__ float gs[1024];
    const int id = blockIdx.x * 256 + threadIdx.x;   // 0..4095
    const int b = id >> 9, o = id & 511;
    const float* gb = g + b * 1024;
    for (int i = threadIdx.x; i < 1024; i += 256) gs[i] = gb[i];
    __syncthreads();
    float a = bc1[o];
    const float* w = Wc1 + (long)o * 1536;
    for (int c = 0; c < 1024; c++) a += w[c] * gs[c];
    ghb[id] = a;
}

// Final 13-class head: out = Wc3 @ h2 + bc3
__launch_bounds__(256)
__global__ void head_kernel(const float* __restrict__ h2, const float* __restrict__ Wc3,
                            const float* __restrict__ bc3, float* __restrict__ out)
{
    __shared__ float Ws[13 * 256];
    __shared__ float bs[13];
    for (int j = threadIdx.x; j < 13 * 256; j += 256) Ws[j] = Wc3[j];
    if (threadIdx.x < 13) bs[threadIdx.x] = bc3[threadIdx.x];
    __syncthreads();
    long i = (long)blockIdx.x * 256 + threadIdx.x;
    if (i >= (long)BB * 13 * NN) return;
    int n = (int)(i % NN); long t = i / NN; int o = (int)(t % 13); int b = (int)(t / 13);
    float a = bs[o];
    const float* hp = h2 + (long)b * 256 * NN + n;
    const float* w = &Ws[o * 256];
    for (int c = 0; c < 256; c++) a += w[c] * hp[(long)c * NN];
    out[i] = a;
}

// ---------------------------------------------------------------------------
static void launch_gemm(hipStream_t s, const float* A, long sA, int lda,
                        const float* B, long sB, int ldb,
                        float* C, long sC, int ldc,
                        const float* bias, long sBias, int M, int Nc, int Kc)
{
    dim3 grid(Nc / 128, M / 128, BB);
    gemm_wmma_f16<<<grid, 256, 0, s>>>(A, sA, lda, B, sB, ldb, C, sC, ldc,
                                       bias, sBias, M, Nc, Kc);
}

extern "C" void kernel_launch(void* const* d_in, const int* in_sizes, int n_in,
                              void* d_out, int out_size, void* d_ws, size_t ws_size,
                              hipStream_t stream)
{
    (void)in_sizes; (void)n_in; (void)out_size; (void)ws_size;
    const float* x   = (const float*)d_in[0];
    const float* We1 = (const float*)d_in[1];
    const float* be1 = (const float*)d_in[2];
    const float* We2 = (const float*)d_in[3];
    const float* be2 = (const float*)d_in[4];
    const float* Wq  = (const float*)d_in[5];
    const float* Wk  = (const float*)d_in[6];
    const float* Wt  = (const float*)d_in[7];
    const float* bt  = (const float*)d_in[8];
    const float* Wf  = (const float*)d_in[9];
    const float* bf  = (const float*)d_in[10];
    const float* Wc1 = (const float*)d_in[11];
    const float* bc1 = (const float*)d_in[12];
    const float* Wc2 = (const float*)d_in[13];
    const float* bc2 = (const float*)d_in[14];
    const float* Wc3 = (const float*)d_in[15];
    const float* bc3 = (const float*)d_in[16];
    float* out = (float*)d_out;

    // -------- workspace layout (float-element offsets) --------
    float* ws = (float*)d_ws;
    size_t o = 0;
    int*   idx   = (int*)(ws + o);        o += (size_t)BB * NN * KNN;      // 0.5M
    float* f1    = ws + o;                o += (size_t)BB * 64 * NN;       // 1M
    float* f2    = ws + o;                o += (size_t)BB * 64 * NN;       // 1M
    float* feat  = ws + o;                o += (size_t)BB * 128 * NN;      // 2M
    float* cat   = ws + o;                o += (size_t)BB * 512 * NN;      // 8M
    float* xq    = ws + o;                o += (size_t)BB * NN * 32;       // 0.5M
    float* xk    = ws + o;                o += (size_t)BB * 32 * NN;       // 0.5M
    float* xr    = ws + o;                o += (size_t)BB * 128 * NN;      // 2M
    float* tbuf  = ws + o;                o += (size_t)BB * 128 * NN;      // 2M
    float* attn  = ws + o;                o += (size_t)BB * NN * NN;       // 33.5M (also fuse y)
    float* h1    = ws + o;                o += (size_t)BB * 512 * NN;      // 8M
    float* h2    = ws + o;                o += (size_t)BB * 256 * NN;      // 4M
    float* epart = ws + o;                o += (size_t)2 * BB * NN * 64;   // 2M
    float* ssum  = ws + o;                o += 1024;
    float* ssq   = ws + o;                o += 1024;
    float* psum  = ws + o;                o += (size_t)BB * 1024;
    float* psq   = ws + o;                o += (size_t)BB * 1024;
    float* pmax  = ws + o;                o += (size_t)BB * 1024;
    float* gbuf  = ws + o;                o += (size_t)BB * 1024;
    float* ghb   = ws + o;                o += (size_t)BB * 512;
    float* ep1 = epart;
    float* ep2 = epart + (size_t)BB * NN * 64;

    const float invBNK = 1.f / ((float)BB * NN * KNN);
    const float invBN  = 1.f / ((float)BB * NN);

    // -------- EdgeConv block 1: xyz -> 64 --------
    knn_kernel<<<BB * NN, 256, 0, stream>>>(x, 3, idx);
    edge1_kernel<<<BB * NN, 256, 0, stream>>>(x, idx, We1, be1, f1, ep1, ep2);
    epart_reduce_kernel<<<64, 256, 0, stream>>>(ep1, ep2, ssum, ssq);
    bn_relu_kernel<<<(BB * 64 * NN) / 256, 256, 0, stream>>>(f1, ssum, ssq, 64, invBNK,
                                                             (long)BB * 64 * NN);
    // -------- EdgeConv block 2: 64 -> 64 --------
    knn_kernel<<<BB * NN, 256, 0, stream>>>(f1, 64, idx);
    edge2_kernel<<<BB * NN, 256, 0, stream>>>(f1, idx, We2, be2, f2, ep1, ep2);
    epart_reduce_kernel<<<64, 256, 0, stream>>>(ep1, ep2, ssum, ssq);
    bn_relu_kernel<<<(BB * 64 * NN) / 256, 256, 0, stream>>>(f2, ssum, ssq, 64, invBNK,
                                                             (long)BB * 64 * NN);
    concat_kernel<<<(BB * 128 * NN) / 256, 256, 0, stream>>>(f1, f2, feat,
                                                             (long)BB * 128 * NN);

    // -------- 4 stacked single-head SA layers --------
    const float* xs = feat;
    long xsStride = 128L * NN;
    for (int i = 0; i < 4; i++) {
        qk_kernel<<<BB * (NN / 8), 256, 0, stream>>>(xs, xsStride,
                                                     Wq + (long)i * 32 * 128,
                                                     Wk + (long)i * 32 * 128, xq, xk);
        // energy[b] = xq[b] (2048x32) @ xk[b] (32x2048)  -> attn
        launch_gemm(stream, xq, (long)NN * 32, 32, xk, 32L * NN, NN,
                    attn, (long)NN * NN, NN, nullptr, 0, NN, NN, 32);
        softmax_kernel<<<BB * NN, 256, 0, stream>>>(attn);
        colscale_kernel<<<BB * 8, 256, 0, stream>>>(attn);
        // xr[b] = x[b] (128x2048) @ attn[b] (2048x2048)
        launch_gemm(stream, xs, xsStride, NN, attn, (long)NN * NN, NN,
                    xr, 128L * NN, NN, nullptr, 0, 128, NN, NN);
        sub_kernel<<<(BB * 128 * NN) / 256, 256, 0, stream>>>(xs, xsStride, xr,
                                                              (long)BB * 128 * NN);
        // t[b] = Wt_i (128x128) @ u[b] + bt_i
        launch_gemm(stream, Wt + (long)i * 128 * 128, 0, 128, xr, 128L * NN, NN,
                    tbuf, 128L * NN, NN, bt + (long)i * 128, 0, 128, NN, 128);
        chan_stats_kernel<<<128, 256, 0, stream>>>(tbuf, 128, ssum, ssq);
        sa_out_kernel<<<(BB * 128 * NN) / 256, 256, 0, stream>>>(
            xs, xsStride, tbuf, ssum, ssq, invBN, cat + (long)i * 128 * NN,
            (long)BB * 128 * NN);
        xs = cat + (long)i * 128 * NN;
        xsStride = 512L * NN;
    }

    // -------- fuse: y = Wf @ cat + bf ; g = relu(BN(y)).max(n)  --------
    float* y = attn;  // reuse attention scratch (16.8M floats <= 33.5M)
    launch_gemm(stream, Wf, 0, 512, cat, 512L * NN, NN,
                y, 1024L * NN, NN, bf, 0, 1024, NN, 512);
    ystats_kernel<<<BB * 1024, 256, 0, stream>>>(y, psum, psq, pmax);
    g_kernel<<<4, 256, 0, stream>>>(psum, psq, pmax, gbuf);
    gh_kernel<<<16, 256, 0, stream>>>(Wc1, bc1, gbuf, ghb);

    // -------- h1 = relu(BN(Wc1[:,1024:] @ cat + (Wc1[:,:1024]@g + bc1))) --------
    launch_gemm(stream, Wc1 + 1024, 0, 1536, cat, 512L * NN, NN,
                h1, 512L * NN, NN, ghb, 512, 512, NN, 512);
    chan_stats_kernel<<<512, 256, 0, stream>>>(h1, 512, ssum, ssq);
    bn_relu_kernel<<<(BB * 512 * NN) / 256, 256, 0, stream>>>(h1, ssum, ssq, 512, invBN,
                                                              (long)BB * 512 * NN);
    // -------- h2 = relu(BN(Wc2 @ h1 + bc2)) --------
    launch_gemm(stream, Wc2, 0, 512, h1, 512L * NN, NN,
                h2, 256L * NN, NN, bc2, 0, 256, NN, 512);
    chan_stats_kernel<<<256, 256, 0, stream>>>(h2, 256, ssum, ssq);
    bn_relu_kernel<<<(BB * 256 * NN) / 256, 256, 0, stream>>>(h2, ssum, ssq, 256, invBN,
                                                              (long)BB * 256 * NN);
    // -------- out = Wc3 @ h2 + bc3 --------
    head_kernel<<<(BB * 13 * NN) / 256, 256, 0, stream>>>(h2, Wc3, bc3, out);
}
